// SeqTrackDecoderXL_11106785428088
// MI455X (gfx1250) — matrix operand order repaired and checked by reading, not compiled, not run on hardware
//
#include <hip/hip_runtime.h>
#include <hip/hip_bf16.h>

typedef __attribute__((ext_vector_type(16))) _Float16 v16h;
typedef __attribute__((ext_vector_type(8)))  float    v8f;

#define QLEN   1024
#define BSZ    8
#define DMODEL 512
#define NHEAD  8
#define DHEAD  64
#define N3     1536
#define LN_EPS 1e-5f

static __device__ __forceinline__ v8f wmma_f16(v16h a, v16h b, v8f c) {
  return __builtin_amdgcn_wmma_f32_16x16x32_f16(false, a, false, b, (short)0, c, false, false);
}

// A-operand (16x32 f16, M x K): lane<16 -> M=lane, K in {kk+0..15}
//                               lane>=16 -> M=lane-16, K in {kk+16..31}
static __device__ __forceinline__ v16h load_a16(const _Float16* A, int lda, int row0, int kk, int lane) {
  int r  = row0 + (lane & 15);
  int kb = kk + ((lane & 16) ? 16 : 0);
  const _Float16* p = A + (size_t)r * lda + kb;
  v16h out;
#pragma unroll
  for (int h = 0; h < 8; ++h) { out[h] = p[h]; out[h + 8] = p[8 + h]; }
  return out;
}

// B-operand: per-lane 16 contiguous f16 (lane = N column, source K-contiguous)
static __device__ __forceinline__ v16h load_row16(const _Float16* p) {
  v16h out;
#pragma unroll
  for (int h = 0; h < 16; ++h) out[h] = p[h];
  return out;
}

// ---------------- converts ----------------
__global__ void k_cvt(const float* __restrict__ src, _Float16* __restrict__ dst, int n) {
  int i = blockIdx.x * 256 + threadIdx.x;
  if (i < n) dst[i] = (_Float16)src[i];
}

// transpose-convert: dst[n*K + k] = src[k*N + n]  (weights become [N][K] f16)
__global__ void k_cvt_t(const float* __restrict__ src, _Float16* __restrict__ dst, int K, int N) {
  int i = blockIdx.x * 256 + threadIdx.x;
  if (i < K * N) {
    int n = i / K, k = i - n * K;
    dst[i] = (_Float16)src[(size_t)k * N + n];
  }
}

__global__ void k_slice_r(const float* __restrict__ r, _Float16* __restrict__ dst) {
  int i = blockIdx.x * 256 + threadIdx.x;        // 1024*512 elements
  int q = i >> 9, c = i & 511;
  dst[i] = (_Float16)r[(size_t)q * (BSZ * DMODEL) + c];   // r[:,0,:]
}

// ---------------- QKV projection ----------------
// X[8192,512] @ Wqkv; WT is Wqkv^T [1536][512] so B loads are K-contiguous.
// Epilogue scatters heads:
//   part0 -> Q1 = Q + r_w_bias, Q2 = Q + r_r_bias   [b][n][q][64]
//   part1 -> K                                      [b][n][q][64]
//   part2 -> Vt (transposed)                        [b][n][64][q]
__global__ __launch_bounds__(256) void k_gemm_qkv(
    const _Float16* __restrict__ X, const _Float16* __restrict__ WT,
    const float* __restrict__ rwb, const float* __restrict__ rrb,
    _Float16* __restrict__ Q1, _Float16* __restrict__ Q2,
    _Float16* __restrict__ K, _Float16* __restrict__ Vt) {
  int lane = threadIdx.x & 31, w = threadIdx.x >> 5;
  int i0 = blockIdx.x * 16;
  int n0 = blockIdx.y * 128;
  int nn = n0 + w * 16 + (lane & 15);
  int hb = (lane & 16) ? 16 : 0;
  const _Float16* bt = WT + (size_t)nn * DMODEL + hb;
  v8f acc = {};
  for (int kt = 0; kt < 16; ++kt) {
    int kk = kt * 32;
    if (kt + 1 < 16) __builtin_prefetch(bt + kk + 32, 0, 0);
    v16h a = load_a16(X, DMODEL, i0, kk, lane);
    v16h b = load_row16(bt + kk);
    acc = wmma_f16(a, b, acc);
  }
  int part = nn >> 9;
  int hc = nn & 511;
  int n = hc >> 6, d = hc & 63;
#pragma unroll
  for (int v = 0; v < 8; ++v) {
    int i = i0 + v + ((lane & 16) ? 8 : 0);
    int q = i >> 3, b7 = i & 7;
    float val = acc[v];
    size_t hidx = (size_t)((b7 * NHEAD + n) * QLEN + q) * DHEAD + d;
    if (part == 0) {
      Q1[hidx] = (_Float16)(val + rwb[n * DHEAD + d]);
      Q2[hidx] = (_Float16)(val + rrb[n * DHEAD + d]);
    } else if (part == 1) {
      K[hidx] = (_Float16)val;
    } else {
      Vt[(size_t)((b7 * NHEAD + n) * DHEAD + d) * QLEN + q] = (_Float16)val;
    }
  }
}

// ---------------- R projection ----------------
// Rh[1024,512] @ Wr (WT = Wr^T [512][512]) -> Rhead [n][q][64]
__global__ __launch_bounds__(256) void k_gemm_r(
    const _Float16* __restrict__ R, const _Float16* __restrict__ WT,
    _Float16* __restrict__ Rhead) {
  int lane = threadIdx.x & 31, w = threadIdx.x >> 5;
  int i0 = blockIdx.x * 16;
  int n0 = blockIdx.y * 128;
  int nn = n0 + w * 16 + (lane & 15);
  int hb = (lane & 16) ? 16 : 0;
  const _Float16* bt = WT + (size_t)nn * DMODEL + hb;
  v8f acc = {};
  for (int kt = 0; kt < 16; ++kt) {
    int kk = kt * 32;
    v16h a = load_a16(R, DMODEL, i0, kk, lane);
    v16h b = load_row16(bt + kk);
    acc = wmma_f16(a, b, acc);
  }
  int n = nn >> 6, d = nn & 63;
#pragma unroll
  for (int v = 0; v < 8; ++v) {
    int i = i0 + v + ((lane & 16) ? 8 : 0);
    Rhead[(size_t)(n * QLEN + i) * DHEAD + d] = (_Float16)acc[v];
  }
}

// ---------------- fused attention ----------------
// Per (bn, 16-query tile): AC = Q1 K^T, S = Q2 Rrev^T (rel-shift via gather S[i,i-j]),
// causal mask + softmax, then P @ V via WMMA. LDS: SC 64KB + SB 64KB + P 32KB = 160KB.
__global__ __launch_bounds__(256) void k_attn(
    const _Float16* __restrict__ Q1, const _Float16* __restrict__ Q2,
    const _Float16* __restrict__ Kh, const _Float16* __restrict__ Vt,
    const _Float16* __restrict__ Rhead, _Float16* __restrict__ AV) {
  extern __shared__ char smem[];
  float*    SC = (float*)smem;                      // scores / exp      [16][1024]
  float*    SB = SC + 16 * 1024;                    // shifted-BD (S)    [16][1024]
  _Float16* P  = (_Float16*)(SB + 16 * 1024);       // probabilities f16 [16][1024]
  int lane = threadIdx.x & 31, w = threadIdx.x >> 5;
  int qb = blockIdx.x * 16;
  int bn = blockIdx.y;
  int n = bn & 7;
  const _Float16* q1 = Q1 + (size_t)bn * QLEN * DHEAD;
  const _Float16* q2 = Q2 + (size_t)bn * QLEN * DHEAD;
  const _Float16* kh = Kh + (size_t)bn * QLEN * DHEAD;
  const _Float16* vt = Vt + (size_t)bn * DHEAD * QLEN;
  const _Float16* rh = Rhead + (size_t)n * QLEN * DHEAD;

  // Phase A+B: strips of AC and S across all 64 key tiles
  for (int jt = w; jt < 64; jt += 8) {
    int j0 = jt * 16;
    int hb = (lane & 16) ? 16 : 0;
    v8f accS = {}, accA = {};
#pragma unroll
    for (int kq = 0; kq < 2; ++kq) {
      int kk = kq * 32;
      v16h a2 = load_a16(q2, DHEAD, qb, kk, lane);
      int rrow = (QLEN - 1) - (j0 + (lane & 15));           // reversed R rows
      v16h bS = load_row16(rh + (size_t)rrow * DHEAD + kk + hb);
      accS = wmma_f16(a2, bS, accS);
      v16h a1 = load_a16(q1, DHEAD, qb, kk, lane);
      v16h bA = load_row16(kh + (size_t)(j0 + (lane & 15)) * DHEAD + kk + hb);
      accA = wmma_f16(a1, bA, accA);
    }
#pragma unroll
    for (int v = 0; v < 8; ++v) {
      int rl = v + ((lane & 16) ? 8 : 0);
      int c = j0 + (lane & 15);
      SB[rl * 1024 + c] = accS[v];
      SC[rl * 1024 + c] = accA[v];
    }
  }
  __syncthreads();

  // Phase C: combine + causal mask + softmax, 16 threads per query row
  {
    int rl = threadIdx.x >> 4;
    int c0 = threadIdx.x & 15;
    int i = qb + rl;
    float mx = -1e30f;
    for (int j = c0; j < 1024; j += 16) {
      float sc = -1e30f;
      if (j <= i) sc = (SC[rl * 1024 + j] + SB[rl * 1024 + (i - j)]) * 0.125f;
      SC[rl * 1024 + j] = sc;
      mx = fmaxf(mx, sc);
    }
#pragma unroll
    for (int m = 1; m < 16; m <<= 1) mx = fmaxf(mx, __shfl_xor(mx, m, 16));
    float sum = 0.f;
    for (int j = c0; j < 1024; j += 16) {
      float e = (j <= i) ? __expf(SC[rl * 1024 + j] - mx) : 0.f;
      SC[rl * 1024 + j] = e;
      sum += e;
    }
#pragma unroll
    for (int m = 1; m < 16; m <<= 1) sum += __shfl_xor(sum, m, 16);
    float inv = 1.f / sum;
    for (int j = c0; j < 1024; j += 16)
      P[rl * 1024 + j] = (_Float16)(SC[rl * 1024 + j] * inv);
  }
  __syncthreads();

  // Phase D: attn_vec = P[16,1024] @ V[1024,64]; waves split (4 n-tiles) x (2 K halves)
  int nt = w & 3, khalf = w >> 2;
  int d0 = nt * 16;
  int hb = (lane & 16) ? 16 : 0;
  v8f acc = {};
  for (int kk = khalf * 512; kk < khalf * 512 + 512; kk += 32) {
    v16h a = load_a16((const _Float16*)P, 1024, 0, kk, lane);
    v16h b = load_row16(vt + (size_t)(d0 + (lane & 15)) * QLEN + kk + hb);
    acc = wmma_f16(a, b, acc);
  }
  float* part = SC;  // reuse
  if (w >= 4) {
#pragma unroll
    for (int v = 0; v < 8; ++v) {
      int rl = v + ((lane & 16) ? 8 : 0);
      part[((w - 4) * 16 + rl) * 16 + (lane & 15)] = acc[v];
    }
  }
  __syncthreads();
  if (w < 4) {
    int b7 = bn >> 3;
#pragma unroll
    for (int v = 0; v < 8; ++v) {
      int rl = v + ((lane & 16) ? 8 : 0);
      float val = acc[v] + part[(w * 16 + rl) * 16 + (lane & 15)];
      int qrow = qb + rl;
      AV[(size_t)(qrow * BSZ + b7) * DMODEL + n * DHEAD + d0 + (lane & 15)] = (_Float16)val;
    }
  }
}

// ---------------- output projection + residual + LayerNorm ----------------
// One WG covers 16 rows x all 512 cols so LN stats reduce locally.
// WT = Wo^T [512][512] f16 -> B loads are contiguous b128s, no staging needed.
__global__ __launch_bounds__(256) void k_out_ln(
    const _Float16* __restrict__ AV, const _Float16* __restrict__ WT,
    const float* __restrict__ wres, const float* __restrict__ g,
    const float* __restrict__ bb, float* __restrict__ out) {
  __shared__ float red[16][2];
  int lane = threadIdx.x & 31, w = threadIdx.x >> 5;
  int i0 = blockIdx.x * 16;
  int hb = (lane & 16) ? 16 : 0;
  v8f acc[4];
#pragma unroll
  for (int j = 0; j < 4; ++j) acc[j] = (v8f){};
  for (int kt = 0; kt < 16; ++kt) {
    int kk = kt * 32;
    v16h a = load_a16(AV, DMODEL, i0, kk, lane);
#pragma unroll
    for (int j = 0; j < 4; ++j) {
      int nn = (w + 8 * j) * 16 + (lane & 15);
      v16h b = load_row16(WT + (size_t)nn * DMODEL + kk + hb);
      acc[j] = wmma_f16(a, b, acc[j]);
    }
  }
  if (threadIdx.x < 16) { red[threadIdx.x][0] = 0.f; red[threadIdx.x][1] = 0.f; }
  __syncthreads();
  float vals[4][8];
#pragma unroll
  for (int j = 0; j < 4; ++j) {
    int c = (w + 8 * j) * 16 + (lane & 15);
#pragma unroll
    for (int v = 0; v < 8; ++v) {
      int rl = v + ((lane & 16) ? 8 : 0);
      int i = i0 + rl;
      float val = acc[j][v] + wres[(size_t)i * DMODEL + c];
      vals[j][v] = val;
      atomicAdd(&red[rl][0], val);
      atomicAdd(&red[rl][1], val * val);
    }
  }
  __syncthreads();
#pragma unroll
  for (int j = 0; j < 4; ++j) {
    int c = (w + 8 * j) * 16 + (lane & 15);
#pragma unroll
    for (int v = 0; v < 8; ++v) {
      int rl = v + ((lane & 16) ? 8 : 0);
      int i = i0 + rl;
      float mu = red[rl][0] * (1.f / 512.f);
      float var = red[rl][1] * (1.f / 512.f) - mu * mu;
      float rs = rsqrtf(var + LN_EPS);
      out[(size_t)i * DMODEL + c] = (vals[j][v] - mu) * rs * g[c] + bb[c];
    }
  }
}

extern "C" void kernel_launch(void* const* d_in, const int* in_sizes, int n_in,
                              void* d_out, int out_size, void* d_ws, size_t ws_size,
                              hipStream_t stream) {
  (void)in_sizes; (void)n_in; (void)out_size; (void)ws_size;
  const float* w    = (const float*)d_in[0];
  const float* r    = (const float*)d_in[1];
  const float* rwb  = (const float*)d_in[2];
  const float* rrb  = (const float*)d_in[3];
  // d_in[4] attn_mask: causal, applied analytically
  const float* Wqkv = (const float*)d_in[5];
  const float* Wr   = (const float*)d_in[6];
  const float* Wo   = (const float*)d_in[7];
  const float* ln_g = (const float*)d_in[8];
  const float* ln_b = (const float*)d_in[9];
  float* out = (float*)d_out;

  char* p = (char*)d_ws;
  auto alloc = [&](size_t bytes) -> char* {
    char* q = p; p += (bytes + 255) & ~(size_t)255; return q;
  };
  const size_t HEADS = (size_t)BSZ * NHEAD * QLEN * DHEAD;   // 4,194,304
  _Float16* Xh     = (_Float16*)alloc((size_t)QLEN * BSZ * DMODEL * 2);
  _Float16* WqkvT  = (_Float16*)alloc((size_t)DMODEL * N3 * 2);
  _Float16* WrT    = (_Float16*)alloc((size_t)DMODEL * DMODEL * 2);
  _Float16* WoT    = (_Float16*)alloc((size_t)DMODEL * DMODEL * 2);
  _Float16* Rh     = (_Float16*)alloc((size_t)QLEN * DMODEL * 2);
  _Float16* Q1     = (_Float16*)alloc(HEADS * 2);
  _Float16* Q2     = (_Float16*)alloc(HEADS * 2);
  _Float16* Kh     = (_Float16*)alloc(HEADS * 2);
  _Float16* Vt     = (_Float16*)alloc(HEADS * 2);
  _Float16* Rhead  = (_Float16*)alloc((size_t)NHEAD * QLEN * DHEAD * 2);
  _Float16* AV     = (_Float16*)alloc((size_t)QLEN * BSZ * DMODEL * 2);

  // converts (weights transposed so WMMA B-operands are K-contiguous)
  k_cvt<<<(QLEN * BSZ * DMODEL) / 256, 256, 0, stream>>>(w, Xh, QLEN * BSZ * DMODEL);
  k_cvt_t<<<(DMODEL * N3) / 256, 256, 0, stream>>>(Wqkv, WqkvT, DMODEL, N3);
  k_cvt_t<<<(DMODEL * DMODEL) / 256, 256, 0, stream>>>(Wr, WrT, DMODEL, DMODEL);
  k_cvt_t<<<(DMODEL * DMODEL) / 256, 256, 0, stream>>>(Wo, WoT, DMODEL, DMODEL);
  k_slice_r<<<(QLEN * DMODEL) / 256, 256, 0, stream>>>(r, Rh);

  // projections
  k_gemm_qkv<<<dim3(QLEN * BSZ / 16, N3 / 128), 256, 0, stream>>>(
      Xh, WqkvT, rwb, rrb, Q1, Q2, Kh, Vt);
  k_gemm_r<<<dim3(QLEN / 16, DMODEL / 128), 256, 0, stream>>>(Rh, WrT, Rhead);

  // fused attention (160KB dynamic LDS)
  int attn_lds = (16 * 1024 * 4) * 2 + 16 * 1024 * 2;
  hipFuncSetAttribute(reinterpret_cast<const void*>(k_attn),
                      hipFuncAttributeMaxDynamicSharedMemorySize, attn_lds);
  k_attn<<<dim3(QLEN / 16, BSZ * NHEAD), 256, attn_lds, stream>>>(
      Q1, Q2, Kh, Vt, Rhead, AV);

  // output projection + residual + LN
  k_out_ln<<<QLEN * BSZ / 16, 256, 0, stream>>>(AV, WoT, w, ln_g, ln_b, out);
}
